// MultiHeadAttention_39067022524523
// MI455X (gfx1250) — matrix operand order repaired and checked by reading, not compiled, not run on hardware
//
#include <hip/hip_runtime.h>
#include <hip/hip_bf16.h>

typedef __attribute__((ext_vector_type(16))) __bf16 v16bf;
typedef __attribute__((ext_vector_type(8)))  __bf16 v8bf;
typedef __attribute__((ext_vector_type(8)))  float  v8f;

static constexpr int Bn = 4, Sn = 1024, En = 1024, NHn = 16, HDn = 64;
// fold softmax scale (1/sqrt(HD)) and log2(e) into Q so we can use exp2 everywhere
static constexpr float QSCALE = 0.125f * 1.4426950408889634f;

#if defined(__HIP_DEVICE_COMPILE__) && \
    __has_builtin(__builtin_amdgcn_global_load_async_to_lds_b128) && \
    __has_builtin(__builtin_amdgcn_s_wait_asynccnt)
#define HAVE_ASYNC_LDS 1
#else
#define HAVE_ASYNC_LDS 0
#endif

#if HAVE_ASYNC_LDS
// exact pointee type per hipcc diagnostic: int __attribute__((vector_size(16)))
typedef int v4i_vs __attribute__((vector_size(16)));
typedef __attribute__((address_space(1))) v4i_vs* g_v4i_p;
typedef __attribute__((address_space(3))) v4i_vs* l_v4i_p;
#endif

static __device__ __forceinline__ __bf16 to_bf16(float f) {
  unsigned u = __builtin_bit_cast(unsigned, f);
  u += 0x7FFFu + ((u >> 16) & 1u);                 // round-to-nearest-even
  unsigned short h = (unsigned short)(u >> 16);
  return __builtin_bit_cast(__bf16, h);
}

static __device__ __forceinline__ v8f vzero() {
  v8f z = {0.f, 0.f, 0.f, 0.f, 0.f, 0.f, 0.f, 0.f};
  return z;
}

// A-fragment (16x32 bf16, row-major source with leading dim ld):
// lanes 0-15: row=row0+l,   K = k0+0..7  and k0+16..23
// lanes16-31: row=row0+l-16,K = k0+8..15 and k0+24..31
static __device__ __forceinline__ v16bf a_frag(const __bf16* Abase, int ld, int row0, int k0) {
  const int l = threadIdx.x & 31;
  const int r = row0 + (l & 15);
  const __bf16* p = Abase + (size_t)r * ld + k0 + ((l >> 4) << 3);
  v8bf lo = *(const v8bf*)p;
  v8bf hi = *(const v8bf*)(p + 16);
  return __builtin_shufflevector(lo, hi, 0,1,2,3,4,5,6,7,8,9,10,11,12,13,14,15);
}

// B-fragment (32x16 bf16) from a pre-transposed Bt [N][K] row-major:
// lanes 0-15: N=n0+l, K=k0+0..7 and k0+16..23 ; lanes 16-31: N=n0+l-16, K=k0+8..15 and k0+24..31
static __device__ __forceinline__ v16bf b_frag(const __bf16* Btbase, int ld, int n0, int k0) {
  const int l = threadIdx.x & 31;
  const int r = n0 + (l & 15);
  const __bf16* p = Btbase + (size_t)r * ld + k0 + ((l >> 4) << 3);
  v8bf lo = *(const v8bf*)p;
  v8bf hi = *(const v8bf*)(p + 16);
  return __builtin_shufflevector(lo, hi, 0,1,2,3,4,5,6,7,8,9,10,11,12,13,14,15);
}

static __device__ __forceinline__ v8f wmma_bf16(v16bf a, v16bf b, v8f c) {
  return __builtin_amdgcn_wmma_f32_16x16x32_bf16(false, a, false, b, (short)0, c, false, false);
}

// xor-shuffle within the wave via ds_swizzle (and_mask=0x1f, xor in [14:10])
template <int M>
static __device__ __forceinline__ float swz_xor(float v) {
  int r = __builtin_amdgcn_ds_swizzle(__builtin_bit_cast(int, v), (M << 10) | 0x1F);
  return __builtin_bit_cast(float, r);
}
static __device__ __forceinline__ float redmax16(float v) {
  v = fmaxf(v, swz_xor<1>(v)); v = fmaxf(v, swz_xor<2>(v));
  v = fmaxf(v, swz_xor<4>(v)); v = fmaxf(v, swz_xor<8>(v));
  return v;
}
static __device__ __forceinline__ float redsum16(float v) {
  v += swz_xor<1>(v); v += swz_xor<2>(v);
  v += swz_xor<4>(v); v += swz_xor<8>(v);
  return v;
}

// 16-byte global->LDS copy: async path when available, sync fallback else.
static __device__ __forceinline__ void cp16_g2l(const __bf16* g, __bf16* l) {
#if HAVE_ASYNC_LDS
  __builtin_amdgcn_global_load_async_to_lds_b128((g_v4i_p)g, (l_v4i_p)l, 0, 0);
#else
  *(v8bf*)l = *(const v8bf*)g;
#endif
}

// ---------------------------------------------------------------- prep kernels
__global__ __launch_bounds__(256) void k_cvt_bf16(const float* __restrict__ in,
                                                  __bf16* __restrict__ out, int n) {
  int i = blockIdx.x * 256 + threadIdx.x;
  if (i < n) out[i] = to_bf16(in[i]);
}

// wt layout: [3][NH][HD][E]; wt[((mat*NH+h)*HD+d)*E+e] = Wmat[h][e][d]
__global__ __launch_bounds__(256) void k_prep_wqkv(const float* __restrict__ Wq,
                                                   const float* __restrict__ Wk,
                                                   const float* __restrict__ Wv,
                                                   __bf16* __restrict__ wt) {
  int i = blockIdx.x * 256 + threadIdx.x;      // exact: 3*NH*HD*E threads
  int e = i & (En - 1);
  int r = i >> 10;
  int d = r & (HDn - 1); r >>= 6;
  int h = r & (NHn - 1);
  int mat = r >> 4;
  const float* W = (mat == 0) ? Wq : (mat == 1) ? Wk : Wv;
  wt[i] = to_bf16(W[((size_t)h * En + e) * HDn + d]);
}

// wot[n][k] = Wo[k][n]
__global__ __launch_bounds__(256) void k_prep_wo(const float* __restrict__ Wo,
                                                 __bf16* __restrict__ wot) {
  int i = blockIdx.x * 256 + threadIdx.x;      // exact: E*E threads
  int k = i & (En - 1);
  int n = i >> 10;
  wot[i] = to_bf16(Wo[(size_t)k * En + n]);
}

// ------------------------------------------------------------ fused QKV GEMM
// Per wave: 32(M) x 64(N) tile of one (b, mat, h) projection; K = E loop of 32.
__global__ __launch_bounds__(256) void k_qkv_gemm(const __bf16* __restrict__ xb,   // [B*S, E]
                                                  const __bf16* __restrict__ wt,   // [3][NH][HD][E]
                                                  const float* __restrict__ bq,
                                                  const float* __restrict__ bk,
                                                  const float* __restrict__ bv,
                                                  __bf16* __restrict__ Q,    // [B*NH][S][HD], pre-scaled
                                                  __bf16* __restrict__ Kb,   // [B*NH][S][HD]
                                                  __bf16* __restrict__ Vt)   // [B*NH][HD][S]
{
  const int wave = (blockIdx.x * 256 + threadIdx.x) >> 5;
  const int mt  = wave & 31;          // 32 m-tiles of 32 rows cover S
  int rem = wave >> 5;                // [0, B*3*NH)
  const int h   = rem & (NHn - 1); rem >>= 4;
  const int mat = rem % 3;
  const int b   = rem / 3;
  const int m0  = mt * 32;

  const __bf16* A  = xb + (size_t)b * Sn * En;
  const __bf16* Bt = wt + (size_t)(mat * NHn + h) * HDn * En;

  v8f acc[2][4];
#pragma unroll
  for (int mm = 0; mm < 2; ++mm)
#pragma unroll
    for (int t = 0; t < 4; ++t) acc[mm][t] = vzero();

  for (int k0 = 0; k0 < En; k0 += 32) {
    v16bf a0 = a_frag(A, En, m0,      k0);
    v16bf a1 = a_frag(A, En, m0 + 16, k0);
#pragma unroll
    for (int t = 0; t < 4; ++t) {
      v16bf bf = b_frag(Bt, En, t * 16, k0);
      acc[0][t] = wmma_bf16(a0, bf, acc[0][t]);
      acc[1][t] = wmma_bf16(a1, bf, acc[1][t]);
    }
  }

  const int l = threadIdx.x & 31, half = l >> 4, nlo = l & 15;
  const float* bias = ((mat == 0) ? bq : (mat == 1) ? bk : bv) + h * HDn;
  const float scl = (mat == 0) ? QSCALE : 1.0f;
  const size_t bh = (size_t)b * NHn + h;
#pragma unroll
  for (int mm = 0; mm < 2; ++mm)
#pragma unroll
    for (int t = 0; t < 4; ++t)
#pragma unroll
      for (int i = 0; i < 8; ++i) {
        int m = m0 + mm * 16 + i + 8 * half;   // token index s
        int n = t * 16 + nlo;                  // head-dim index d
        float v = (acc[mm][t][i] + bias[n]) * scl;
        if (mat == 2)      Vt[(bh * HDn + n) * Sn + m] = to_bf16(v);
        else if (mat == 0) Q [(bh * Sn + m) * HDn + n] = to_bf16(v);
        else               Kb[(bh * Sn + m) * HDn + n] = to_bf16(v);
      }
}

// -------------------------------------------------------- flash-ish attention
// block = (b, h, 64 query rows); 4 waves x 16 rows each. K/V panels for each
// 32-key block are staged into double-buffered LDS via async-to-LDS (shared by
// all 4 waves); scores and P*V both through WMMA; P staged via LDS to convert
// C-layout -> A-layout.
__global__ __launch_bounds__(128) void k_attention(const __bf16* __restrict__ Q,
                                                   const __bf16* __restrict__ Kb,
                                                   const __bf16* __restrict__ Vt,
                                                   __bf16* __restrict__ attn)  // [B*S, E]
{
  __shared__ __align__(32) __bf16 Klds[2][32 * 64];   // 2 x 4KB
  __shared__ __align__(32) __bf16 Vlds[2][64 * 32];   // 2 x 4KB
  __shared__ __align__(32) __bf16 Plds[4][16 * 32];   // 4KB

  const int tid = threadIdx.x;
  const int wid = tid >> 5;
  const int l = tid & 31, half = l >> 4, nlo = l & 15;
  const int blk = blockIdx.x;          // B*NH*(S/64)
  const int st = blk & 15;             // 64-row slab index
  const int bh = blk >> 4;             // b*NH + h
  const int m0 = st * 64 + wid * 16;

  const __bf16* q  = Q  + (size_t)bh * Sn * HDn;
  const __bf16* k  = Kb + (size_t)bh * Sn * HDn;
  const __bf16* vt = Vt + (size_t)bh * HDn * Sn;

  // stage K rows [sk, sk+32) (contiguous 4KB) and Vt cols [sk, sk+32) of all 64
  // rows into LDS; 4 x b128 per thread per stage.
  auto stage_kv = [&](int sk, __bf16* kl, __bf16* vl) {
    const __bf16* ksrc = k + (size_t)sk * HDn;
#pragma unroll
    for (int j = 0; j < 2; ++j) {
      int e = (tid + j * 128) * 8;                      // 16B chunks of 2048 elems
      cp16_g2l(ksrc + e, kl + e);
    }
#pragma unroll
    for (int j = 0; j < 2; ++j) {
      int c = tid + j * 128;                            // 256 chunks: row d, 8-col piece
      int row = c >> 2, co = (c & 3) * 8;
      cp16_g2l(vt + (size_t)row * Sn + sk + co, vl + row * 32 + co);
    }
  };

  const v16bf qa0 = a_frag(q, HDn, m0, 0);
  const v16bf qa1 = a_frag(q, HDn, m0, 32);

  v8f acc[4];
#pragma unroll
  for (int t = 0; t < 4; ++t) acc[t] = vzero();
  float mrow[8], lrow[8];
#pragma unroll
  for (int i = 0; i < 8; ++i) { mrow[i] = -3.0e38f; lrow[i] = 0.f; }

  __bf16* pb = Plds[wid];

  stage_kv(0, Klds[0], Vlds[0]);
  int cur = 0;

  for (int sk = 0; sk < Sn; sk += 32) {
    if (sk + 32 < Sn) {
      stage_kv(sk + 32, Klds[cur ^ 1], Vlds[cur ^ 1]);  // prefetch next panel
#if HAVE_ASYNC_LDS
      __builtin_amdgcn_s_wait_asynccnt(4);              // current panel landed,
#endif                                                  // next may stay in flight
    } else {
#if HAVE_ASYNC_LDS
      __builtin_amdgcn_s_wait_asynccnt(0);
#endif
    }
    __syncthreads();                                    // panel visible to all waves
    const __bf16* kl = Klds[cur];
    const __bf16* vl = Vlds[cur];

    // scores tile [16 x 32] (scale already folded into Q, base-2 domain)
    v8f c0 = vzero(), c1 = vzero();
    c0 = wmma_bf16(qa0, b_frag(kl, HDn, 0,  0),  c0);
    c0 = wmma_bf16(qa1, b_frag(kl, HDn, 0,  32), c0);
    c1 = wmma_bf16(qa0, b_frag(kl, HDn, 16, 0),  c1);
    c1 = wmma_bf16(qa1, b_frag(kl, HDn, 16, 32), c1);

    float corr[8];
#pragma unroll
    for (int i = 0; i < 8; ++i) {
      float mx = redmax16(fmaxf(c0[i], c1[i]));
      float mn = fmaxf(mrow[i], mx);
      corr[i] = __builtin_amdgcn_exp2f(mrow[i] - mn);
      mrow[i] = mn;
    }
#pragma unroll
    for (int i = 0; i < 8; ++i) {
      float p0 = __builtin_amdgcn_exp2f(c0[i] - mrow[i]);
      float p1 = __builtin_amdgcn_exp2f(c1[i] - mrow[i]);
      c0[i] = p0; c1[i] = p1;
      lrow[i] = lrow[i] * corr[i] + redsum16(p0 + p1);
    }
#pragma unroll
    for (int t = 0; t < 4; ++t)
#pragma unroll
      for (int i = 0; i < 8; ++i) acc[t][i] *= corr[i];

    // C-layout -> LDS [16 rows][32 cols] bf16 (per-wave buffer, in-wave order)
#pragma unroll
    for (int i = 0; i < 8; ++i) {
      int m = i + 8 * half;
      pb[m * 32 + nlo]      = to_bf16(c0[i]);
      pb[m * 32 + 16 + nlo] = to_bf16(c1[i]);
    }
    v16bf pf = a_frag(pb, 32, 0, 0);   // read back in A-layout (ds_read)

#pragma unroll
    for (int t = 0; t < 4; ++t)
      acc[t] = wmma_bf16(pf, b_frag(vl, 32, t * 16, 0), acc[t]);

    __syncthreads();                   // all waves done with panel before overwrite
    cur ^= 1;
  }

  // normalize + write concat-head layout [B*S, E] as bf16 for final GEMM
  const int bb = bh >> 4, hh = bh & 15;
#pragma unroll
  for (int t = 0; t < 4; ++t)
#pragma unroll
    for (int i = 0; i < 8; ++i) {
      int m = m0 + i + 8 * half;
      int d = t * 16 + nlo;
      float v = acc[t][i] / lrow[i];
      attn[((size_t)bb * Sn + m) * En + hh * HDn + d] = to_bf16(v);
    }
}

// ------------------------------------------------------- output projection
__global__ __launch_bounds__(256) void k_out_gemm(const __bf16* __restrict__ attn, // [B*S, E]
                                                  const __bf16* __restrict__ wot,  // [E][E] = Wo^T
                                                  const float* __restrict__ bo,
                                                  float* __restrict__ out)         // [B*S, E]
{
  const int wave = (blockIdx.x * 256 + threadIdx.x) >> 5;
  const int nt = wave & 15;          // 16 n-tiles of 64 cover E
  const int mt = wave >> 4;          // 128 m-tiles of 32 cover B*S
  const int m0 = mt * 32, n0 = nt * 64;

  v8f acc[2][4];
#pragma unroll
  for (int mm = 0; mm < 2; ++mm)
#pragma unroll
    for (int t = 0; t < 4; ++t) acc[mm][t] = vzero();

  for (int k0 = 0; k0 < En; k0 += 32) {
    v16bf a0 = a_frag(attn, En, m0,      k0);
    v16bf a1 = a_frag(attn, En, m0 + 16, k0);
#pragma unroll
    for (int t = 0; t < 4; ++t) {
      v16bf bf = b_frag(wot, En, n0 + t * 16, k0);
      acc[0][t] = wmma_bf16(a0, bf, acc[0][t]);
      acc[1][t] = wmma_bf16(a1, bf, acc[1][t]);
    }
  }

  const int l = threadIdx.x & 31, half = l >> 4, nlo = l & 15;
#pragma unroll
  for (int mm = 0; mm < 2; ++mm)
#pragma unroll
    for (int t = 0; t < 4; ++t)
#pragma unroll
      for (int i = 0; i < 8; ++i) {
        int m = m0 + mm * 16 + i + 8 * half;
        int n = n0 + t * 16 + nlo;
        out[(size_t)m * En + n] = acc[mm][t][i] + bo[n];
      }
}

// ------------------------------------------------------------------- launch
extern "C" void kernel_launch(void* const* d_in, const int* in_sizes, int n_in,
                              void* d_out, int out_size, void* d_ws, size_t ws_size,
                              hipStream_t stream) {
  const float* x  = (const float*)d_in[0];
  const float* Wq = (const float*)d_in[1];
  const float* bq = (const float*)d_in[2];
  const float* Wk = (const float*)d_in[3];
  const float* bk = (const float*)d_in[4];
  const float* Wv = (const float*)d_in[5];
  const float* bv = (const float*)d_in[6];
  const float* Wo = (const float*)d_in[7];
  const float* bo = (const float*)d_in[8];
  float* out = (float*)d_out;

  char* w = (char*)d_ws;
  auto take = [&](size_t bytes) {
    char* p = w;
    w += (bytes + 255) & ~(size_t)255;
    return p;
  };
  __bf16* xb   = (__bf16*)take((size_t)Bn * Sn * En * 2);          // 8 MB
  __bf16* wt   = (__bf16*)take((size_t)3 * NHn * HDn * En * 2);    // 6 MB
  __bf16* wot  = (__bf16*)take((size_t)En * En * 2);               // 2 MB
  __bf16* Qb   = (__bf16*)take((size_t)Bn * NHn * Sn * HDn * 2);   // 8 MB
  __bf16* Kbuf = (__bf16*)take((size_t)Bn * NHn * Sn * HDn * 2);   // 8 MB
  __bf16* Vt   = (__bf16*)take((size_t)Bn * NHn * HDn * Sn * 2);   // 8 MB
  __bf16* attn = (__bf16*)take((size_t)Bn * Sn * En * 2);          // 8 MB

  const int nX = Bn * Sn * En;                       // 4194304
  k_cvt_bf16<<<nX / 256, 256, 0, stream>>>(x, xb, nX);
  k_prep_wqkv<<<(3 * NHn * HDn * En) / 256, 256, 0, stream>>>(Wq, Wk, Wv, wt);
  k_prep_wo<<<(En * En) / 256, 256, 0, stream>>>(Wo, wot);

  // waves = (S/32) * B*3*NH = 6144 -> 768 blocks of 8 waves
  k_qkv_gemm<<<768, 256, 0, stream>>>(xb, wt, bq, bk, bv, Qb, Kbuf, Vt);

  // blocks = B*NH*(S/64) = 1024, 4 waves each
  k_attention<<<Bn * NHn * (Sn / 64), 128, 0, stream>>>(Qb, Kbuf, Vt, attn);

  // waves = (B*S/32) * (E/64) = 2048 -> 256 blocks
  k_out_gemm<<<256, 256, 0, stream>>>(attn, wot, bo, out);
}
